// ResNetWSL_18485539242484
// MI455X (gfx1250) — matrix operand order repaired and checked by reading, not compiled, not run on hardware
//
#include <hip/hip_runtime.h>

typedef __attribute__((ext_vector_type(16))) _Float16 v16h;
typedef __attribute__((ext_vector_type(8)))  _Float16 v8h;
typedef __attribute__((ext_vector_type(8)))  float    v8f;

#define N_CLS 20
#define N_MAPS 8
#define EMB 512
#define HWSZ 64               // 8*8
#define OCH (N_CLS * N_MAPS)  // 160

// One workgroup per batch element. 256 threads = 8 wave32.
// Fused: downconv GEMM (WMMA f16->f32, A and B both fed from LDS) + GMP +
// class-wise pools + score scale + fmap + weighted GAP + classifier GEMV +
// both softmaxes. ~280 KB LDS (within the 320 KB WGP allocation).
__global__ __launch_bounds__(256) void wsl_fused_kernel(
    const float* __restrict__ x,       // (B, 512, 8, 8)
    const float* __restrict__ w_down,  // (160, 512)
    const float* __restrict__ b_down,  // (160,)
    const float* __restrict__ w_cls,   // (20, 1024)
    const float* __restrict__ b_cls,   // (20,)
    float* __restrict__ out1,          // (B, 20) softmax(scores)
    float* __restrict__ out2,          // (B, 20) softmax(logits)
    float* __restrict__ fms)           // (B, 20, 8, 8) feature_map_split
{
    __shared__ _Float16 sXh[HWSZ][EMB];     // 64 KB, x[b] transposed [hw][c], f16
    __shared__ _Float16 sWh[OCH][EMB];      // 160 KB, w_down as f16, row-major
    __shared__ float    sXC[OCH][HWSZ];     // 40 KB, downconv output
    __shared__ float    sGap1[EMB];         // sum_hw x  (f32, exact)
    __shared__ float    sGap2[EMB];         // sum_hw fmap*x
    __shared__ float    sXG[OCH];           // global max pool
    __shared__ float    sScores[N_CLS];
    __shared__ float    sConv[N_CLS][HWSZ]; // 5 KB, scaled class maps
    __shared__ float    sFmap[HWSZ];
    __shared__ float    sLogits[N_CLS];

    const int b    = blockIdx.x;
    const int t    = threadIdx.x;
    const int wave = t >> 5;
    const int lane = t & 31;
    const int lh   = lane >> 4;   // lane-half selector
    const int l15  = lane & 15;

    const float* xb = x + (size_t)b * EMB * HWSZ;

    // hint the classifier weights toward cache before we need them (phase 7)
    if (t < N_CLS) __builtin_prefetch(w_cls + t * (2 * EMB), 0, 3);

    // ---- Phase 1a: w_down (f32, L2-resident) -> LDS f16, converted ONCE.
    // 160*512 / 4 = 20480 float4 units, 80 per thread, coalesced.
    {
        _Float16* wflat = &sWh[0][0];
        const float4* wp = (const float4*)w_down;
        for (int idx = t; idx < OCH * EMB / 4; idx += 256) {
            float4 v = wp[idx];
            _Float16 h0 = (_Float16)v.x, h1 = (_Float16)v.y;
            _Float16 h2 = (_Float16)v.z, h3 = (_Float16)v.w;
            wflat[idx * 4 + 0] = h0;
            wflat[idx * 4 + 1] = h1;
            wflat[idx * 4 + 2] = h2;
            wflat[idx * 4 + 3] = h3;
        }
    }

    // ---- Phase 1b: x[b] (f32, HBM, read once) -> LDS f16 transposed; row sums
    #pragma unroll
    for (int rr = 0; rr < 2; ++rr) {
        const int c = t * 2 + rr;                       // channel 0..511
        const float4* xp = (const float4*)(xb + c * HWSZ);
        float s = 0.f;
        #pragma unroll
        for (int q = 0; q < 16; ++q) {
            float4 v = xp[q];
            s += v.x + v.y + v.z + v.w;
            const int h = q * 4;
            sXh[h + 0][c] = (_Float16)v.x;
            sXh[h + 1][c] = (_Float16)v.y;
            sXh[h + 2][c] = (_Float16)v.z;
            sXh[h + 3][c] = (_Float16)v.w;
        }
        sGap1[c] = s;
    }
    __syncthreads();

    // ---- Phase 2: xc = w_down (160x512) @ x[b] (512x64) + b_down, via WMMA.
    // 40 tiles of 16x16; 8 waves x 5 tiles; K = 512 = 16 steps of 32.
    // Inner loop: 4x ds_load_b128 + 1x v_wmma, no conversions, no global loads.
    for (int i = 0; i < 5; ++i) {
        const int tile = wave * 5 + i;
        const int mo = tile >> 2;       // 0..9
        const int no = tile & 3;        // 0..3
        v8f acc = {};
        for (int ko = 0; ko < 16; ++ko) {
            // A fragment: row M = mo*16 + l15, K = ko*32 + lh*16 + {0..15}
            const _Float16* ap = &sWh[mo * 16 + l15][ko * 32 + lh * 16];
            const v8h alo = *(const v8h*)(ap);
            const v8h ahi = *(const v8h*)(ap + 8);
            v16h a;
            #pragma unroll
            for (int j = 0; j < 8; ++j) { a[j] = alo[j]; a[j + 8] = ahi[j]; }
            // B fragment: column N = no*16 + l15, K = ko*32 + lh*16 + {0..15}
            const v16h bf = *(const v16h*)&sXh[no * 16 + l15][ko * 32 + lh * 16];
            acc = __builtin_amdgcn_wmma_f32_16x16x32_f16(
                false, a, false, bf, (short)0, acc, false, false);
        }
        #pragma unroll
        for (int r = 0; r < 8; ++r) {
            const int M = mo * 16 + lh * 8 + r;
            const int N = no * 16 + l15;
            sXC[M][N] = acc[r] + b_down[M];
        }
    }
    __syncthreads();

    // ---- Phase 3: GMP over 8x8 -> sXG[160]
    if (t < OCH) {
        float m = sXC[t][0];
        #pragma unroll 8
        for (int h = 1; h < HWSZ; ++h) m = fmaxf(m, sXC[t][h]);
        sXG[t] = m;
    }
    __syncthreads();

    // ---- Phase 4: scores = mean over maps -> sScores[20]
    if (t < N_CLS) {
        float s = 0.f;
        #pragma unroll
        for (int m = 0; m < N_MAPS; ++m) s += sXG[t * N_MAPS + m];
        sScores[t] = s * (1.f / N_MAPS);
    }
    __syncthreads();

    // ---- Phase 5: x_conv = mean over maps, scaled by scores -> fms + sConv
    for (int idx = t; idx < N_CLS * HWSZ; idx += 256) {
        const int cl = idx >> 6, h = idx & 63;
        float v = 0.f;
        #pragma unroll
        for (int m = 0; m < N_MAPS; ++m) v += sXC[cl * N_MAPS + m][h];
        v *= (1.f / N_MAPS);
        v *= sScores[cl];
        sConv[cl][h] = v;
        fms[(size_t)b * (N_CLS * HWSZ) + idx] = v;
    }
    __syncthreads();

    // ---- Phase 6: fmap = mean over classes -> sFmap[64]
    if (t < HWSZ) {
        float f = 0.f;
        #pragma unroll
        for (int cl = 0; cl < N_CLS; ++cl) f += sConv[cl][t];
        sFmap[t] = f * (1.f / N_CLS);
    }
    __syncthreads();

    // ---- Phase 7a: gap2[c] = sum_hw fmap[hw] * x[c,hw] (x from f16 LDS copy)
    #pragma unroll
    for (int rr = 0; rr < 2; ++rr) {
        const int c = t * 2 + rr;
        float s = 0.f;
        for (int h = 0; h < HWSZ; ++h) s += sFmap[h] * (float)sXh[h][c];
        sGap2[c] = s;
    }
    __syncthreads();

    // ---- Phase 7b: logits = gap @ w_cls.T + b_cls  (gap = [mean x, mean fmap*x])
    if (t < N_CLS) {
        const float* wr = w_cls + t * (2 * EMB);
        float L = b_cls[t];
        for (int c = 0; c < EMB; ++c) {
            L += (sGap1[c] * (1.f / HWSZ)) * wr[c];
            L += (sGap2[c] * (1.f / HWSZ)) * wr[EMB + c];
        }
        sLogits[t] = L;
    }
    __syncthreads();

    // ---- Phase 8: softmaxes (tiny: 20 elements each)
    if (t == 0) {
        float mx = sScores[0];
        for (int c = 1; c < N_CLS; ++c) mx = fmaxf(mx, sScores[c]);
        float den = 0.f;
        for (int c = 0; c < N_CLS; ++c) den += __expf(sScores[c] - mx);
        const float inv = 1.f / den;
        for (int c = 0; c < N_CLS; ++c)
            out1[(size_t)b * N_CLS + c] = __expf(sScores[c] - mx) * inv;
    } else if (t == 32) {
        float mx = sLogits[0];
        for (int c = 1; c < N_CLS; ++c) mx = fmaxf(mx, sLogits[c]);
        float den = 0.f;
        for (int c = 0; c < N_CLS; ++c) den += __expf(sLogits[c] - mx);
        const float inv = 1.f / den;
        for (int c = 0; c < N_CLS; ++c)
            out2[(size_t)b * N_CLS + c] = __expf(sLogits[c] - mx) * inv;
    }
}

extern "C" void kernel_launch(void* const* d_in, const int* in_sizes, int n_in,
                              void* d_out, int out_size, void* d_ws, size_t ws_size,
                              hipStream_t stream) {
    const float* x      = (const float*)d_in[0];
    const float* w_down = (const float*)d_in[1];
    const float* b_down = (const float*)d_in[2];
    const float* w_cls  = (const float*)d_in[3];
    const float* b_cls  = (const float*)d_in[4];

    const int B = in_sizes[0] / (EMB * HWSZ);   // 512

    float* out  = (float*)d_out;
    float* out1 = out;                          // (B, 20)
    float* out2 = out + (size_t)B * N_CLS;      // (B, 20)
    float* fms  = out + 2 * (size_t)B * N_CLS;  // (B, 20, 8, 8)

    wsl_fused_kernel<<<B, 256, 0, stream>>>(x, w_down, b_down, w_cls, b_cls,
                                            out1, out2, fms);
}
